// VideoAttentionModel_34961033789956
// MI455X (gfx1250) — matrix operand order repaired and checked by reading, not compiled, not run on hardware
//
#include <hip/hip_runtime.h>

typedef __bf16 bf16;
typedef __attribute__((ext_vector_type(16))) __bf16 vbf16x16;
typedef __attribute__((ext_vector_type(8)))  float  v8f;

#define D_MODEL 256
#define SEQ     4096
#define BATCH   8
#define NROWS   (BATCH * SEQ)   // 32768

union ABFrag { vbf16x16 v; uint4 q[2]; unsigned int u[8]; };

// ---------- fragment loaders (ISA 16-bit A 16x32 / B 32x16 layouts, wave32) ----------

// A: 16x32 (MxK). base points at tile element [0][0], row-major, ld elements/row.
// lanes 0-15: M=lane, K groups 0..7 & 16..23 ; lanes 16-31: M=lane-16, K 8..15 & 24..31
__device__ inline vbf16x16 load_frag_a(const bf16* base, int ld) {
  const int lane = threadIdx.x & 31;
  const int h = lane >> 4, m = lane & 15;
  const uint4* r = (const uint4*)(base + (size_t)m * ld);
  ABFrag t;
  t.q[0] = r[h];      // K = h*8 .. h*8+7
  t.q[1] = r[2 + h];  // K = 16 + h*8 .. 16+h*8+7
  return t.v;
}

// B: 32x16 (KxN). Source is "pair-major": rows indexed by output column n,
// columns by K (so pairs along K are contiguous). base at [n=0][k=0].
// lanes 0-15: K 0..7 & 16..23 ; lanes 16-31: K 8..15 & 24..31 (matches A).
__device__ inline vbf16x16 load_frag_b(const bf16* base, int ld) {
  const int lane = threadIdx.x & 31;
  const int h = lane >> 4, n = lane & 15;
  const uint4* r = (const uint4*)(base + (size_t)n * ld);
  ABFrag t;
  t.q[0] = r[h];
  t.q[1] = r[2 + h];
  return t.v;
}

__device__ inline v8f wmma_bf16(vbf16x16 a, vbf16x16 b, v8f c) {
  return __builtin_amdgcn_wmma_f32_16x16x32_bf16(
      /*neg_a=*/false, a, /*neg_b=*/false, b,
      /*c_mod=*/(short)0, c, /*reuse_a=*/false, /*reuse_b=*/false);
}

__device__ inline bf16 f2bf(float f) { return (bf16)f; }

// wave-local LDS RAW fence: slices are wave-private, no block barrier needed
__device__ inline void wave_lds_fence() {
  asm volatile("s_wait_dscnt 0" ::: "memory");
}

// ---------- kernel 1: f32 -> bf16 copy of x ----------
__global__ void convert_x_kernel(const float* __restrict__ x, bf16* __restrict__ xb, int n) {
  int i = blockIdx.x * 256 + threadIdx.x;
  if (i < n) xb[i] = f2bf(x[i]);
}

// ---------- kernel 2: transpose 4 weight matrices to bf16 Wt[n][d] = W[d][n] ----------
__global__ void transpose_w_kernel(const float* __restrict__ Wq, const float* __restrict__ Wk,
                                   const float* __restrict__ Wv, const float* __restrict__ Wo,
                                   bf16* __restrict__ Wqt, bf16* __restrict__ Wkt,
                                   bf16* __restrict__ Wvt, bf16* __restrict__ Wot) {
  int z = blockIdx.y;
  const float* W = (z == 0) ? Wq : (z == 1) ? Wk : (z == 2) ? Wv : Wo;
  bf16* Wt = (z == 0) ? Wqt : (z == 1) ? Wkt : (z == 2) ? Wvt : Wot;
  int i = blockIdx.x * 256 + threadIdx.x;   // 0 .. 65535
  int nrow = i >> 8, d = i & 255;
  Wt[i] = f2bf(W[d * D_MODEL + nrow]);
}

// ---------- kernel 3: QKV projection (z selects Q/K/V). V stored transposed per batch ----------
__global__ void qkv_kernel(const bf16* __restrict__ xb,
                           const bf16* __restrict__ Wqt, const bf16* __restrict__ Wkt,
                           const bf16* __restrict__ Wvt,
                           const float* __restrict__ bq, const float* __restrict__ bk,
                           const float* __restrict__ bv,
                           bf16* __restrict__ Qb, bf16* __restrict__ Kb, bf16* __restrict__ VtB) {
  const int z = blockIdx.z;
  const bf16*  Wt   = (z == 0) ? Wqt : (z == 1) ? Wkt : Wvt;
  const float* bias = (z == 0) ? bq  : (z == 1) ? bk  : bv;
  const int lane = threadIdx.x & 31;
  const int wave = threadIdx.x >> 5;
  const int h = lane >> 4, n = lane & 15;
  const int m0 = blockIdx.x * 128 + wave * 16;         // global row
  const bf16* X = xb + (size_t)m0 * D_MODEL;

  vbf16x16 af[8];
#pragma unroll
  for (int kc = 0; kc < 8; ++kc) af[kc] = load_frag_a(X + kc * 32, D_MODEL);

#pragma unroll 1
  for (int nt = 0; nt < 16; ++nt) {
    // materialize all 8 B fragments first so WMMAs don't stall per-load
    vbf16x16 bf[8];
#pragma unroll
    for (int kc = 0; kc < 8; ++kc)
      bf[kc] = load_frag_b(Wt + nt * 16 * D_MODEL + kc * 32, D_MODEL);
    v8f acc = {};
#pragma unroll
    for (int kc = 0; kc < 8; ++kc) acc = wmma_bf16(af[kc], bf[kc], acc);

    const int col = nt * 16 + n;
    const float bs = bias[col];
    if (z < 2) {
      bf16* dst = (z == 0) ? Qb : Kb;
#pragma unroll
      for (int r = 0; r < 8; ++r)
        dst[(size_t)(m0 + 8 * h + r) * D_MODEL + col] = f2bf(acc[r] + bs);
    } else {
      const int bidx = m0 >> 12;                 // / 4096
      const int s0 = (m0 & 4095) + 8 * h;
      union { uint4 q; bf16 e[8]; } pk;
#pragma unroll
      for (int r = 0; r < 8; ++r) pk.e[r] = f2bf(acc[r] + bs);
      *(uint4*)(VtB + ((size_t)bidx * D_MODEL + col) * SEQ + s0) = pk.q;
    }
  }
}

// ---------- kernel 4: flash attention + fused Wo projection + pooling logits ----------
__global__ void attn_kernel(const bf16* __restrict__ Qb, const bf16* __restrict__ Kb,
                            const bf16* __restrict__ VtB, const bf16* __restrict__ Wot,
                            const float* __restrict__ bo, const float* __restrict__ cv,
                            float* __restrict__ logits, float* __restrict__ wlog) {
  __shared__ __align__(16) bf16 lds[8 * 16 * D_MODEL];   // 64 KB: per-wave 16x256 slice
  const int lane = threadIdx.x & 31;
  const int wave = threadIdx.x >> 5;
  const int h = lane >> 4, n = lane & 15;
  const int bidx = blockIdx.y;
  const int q0 = blockIdx.x * 128 + wave * 16;           // row within batch
  const size_t rowbase = (size_t)bidx * SEQ;

  const bf16* Q  = Qb + (rowbase + q0) * D_MODEL;
  const bf16* K  = Kb + rowbase * D_MODEL;
  const bf16* Vt = VtB + (size_t)bidx * D_MODEL * SEQ;
  bf16* ldsW = lds + wave * (16 * D_MODEL);              // reuse: P (16x32) / ctx (16x256)

  float mrow[8], lrow[8];
  v8f ctx[16];
  v8f vzero = {};
#pragma unroll
  for (int t = 0; t < 16; ++t) ctx[t] = vzero;
#pragma unroll
  for (int r = 0; r < 8; ++r) { mrow[r] = -3.4e38f; lrow[r] = 0.0f; }

  const float scale = 0.0625f;   // 1/sqrt(256)

#pragma unroll 1
  for (int kb = 0; kb < SEQ; kb += 32) {
    // speculative prefetch of next key block (32 rows x 512B, spread across 32 lanes)
    if (kb + 32 < SEQ)
      __builtin_prefetch(K + (size_t)(kb + 32) * D_MODEL + lane * D_MODEL, 0, 0);

    // ---- scores: 16x32 tile = two 16x16 WMMA accumulators over K=256 ----
    // grouped so 12 b128 loads are in flight before each 4-WMMA burst
    v8f s0 = {}, s1 = {};
#pragma unroll
    for (int g = 0; g < 4; ++g) {
      const int kc0 = 2 * g, kc1 = 2 * g + 1;
      vbf16x16 a0  = load_frag_a(Q + kc0 * 32, D_MODEL);
      vbf16x16 a1  = load_frag_a(Q + kc1 * 32, D_MODEL);
      vbf16x16 b00 = load_frag_b(K + (size_t)kb * D_MODEL + kc0 * 32, D_MODEL);
      vbf16x16 b01 = load_frag_b(K + (size_t)(kb + 16) * D_MODEL + kc0 * 32, D_MODEL);
      vbf16x16 b10 = load_frag_b(K + (size_t)kb * D_MODEL + kc1 * 32, D_MODEL);
      vbf16x16 b11 = load_frag_b(K + (size_t)(kb + 16) * D_MODEL + kc1 * 32, D_MODEL);
      s0 = wmma_bf16(a0, b00, s0);
      s1 = wmma_bf16(a0, b01, s1);
      s0 = wmma_bf16(a1, b10, s0);
      s1 = wmma_bf16(a1, b11, s1);
    }
    // ---- online softmax (row = 8h + r, col = n within half; reduce across 16 lanes) ----
    float p0[8], p1[8], corr[8];
#pragma unroll
    for (int r = 0; r < 8; ++r) {
      float x0 = s0[r] * scale, x1 = s1[r] * scale;
      float mx = fmaxf(x0, x1);
      mx = fmaxf(mx, __shfl_xor(mx, 1, 32));
      mx = fmaxf(mx, __shfl_xor(mx, 2, 32));
      mx = fmaxf(mx, __shfl_xor(mx, 4, 32));
      mx = fmaxf(mx, __shfl_xor(mx, 8, 32));
      float mn = fmaxf(mrow[r], mx);
      corr[r] = __expf(mrow[r] - mn);
      p0[r] = __expf(x0 - mn);
      p1[r] = __expf(x1 - mn);
      float rs = p0[r] + p1[r];
      rs += __shfl_xor(rs, 1, 32);
      rs += __shfl_xor(rs, 2, 32);
      rs += __shfl_xor(rs, 4, 32);
      rs += __shfl_xor(rs, 8, 32);
      lrow[r] = lrow[r] * corr[r] + rs;
      mrow[r] = mn;
    }
#pragma unroll
    for (int t = 0; t < 16; ++t)
#pragma unroll
      for (int r = 0; r < 8; ++r) ctx[t][r] *= corr[r];

    // ---- P -> LDS (C/D layout -> row-major 16x32 bf16), then as A-fragment ----
#pragma unroll
    for (int r = 0; r < 8; ++r) {
      ldsW[(8 * h + r) * 32 + n]      = f2bf(p0[r]);
      ldsW[(8 * h + r) * 32 + 16 + n] = f2bf(p1[r]);
    }
    wave_lds_fence();          // slices are wave-private: DS-counter wait only
    vbf16x16 pa = load_frag_a(ldsW, 32);

    // ---- ctx += P (16x32) x V (32x256): 16 WMMAs, B from transposed V, paired loads ----
#pragma unroll
    for (int t = 0; t < 8; ++t) {
      vbf16x16 bva = load_frag_b(Vt + (size_t)(2 * t) * 16 * SEQ + kb, SEQ);
      vbf16x16 bvb = load_frag_b(Vt + (size_t)(2 * t + 1) * 16 * SEQ + kb, SEQ);
      ctx[2 * t]     = wmma_bf16(pa, bva, ctx[2 * t]);
      ctx[2 * t + 1] = wmma_bf16(pa, bvb, ctx[2 * t + 1]);
    }
    wave_lds_fence();          // P reads done before next iteration overwrites
  }

  // ---- finalize ctx = acc / l, stage to LDS as bf16 16x256 ----
  float inv[8];
#pragma unroll
  for (int r = 0; r < 8; ++r) inv[r] = 1.0f / lrow[r];
#pragma unroll
  for (int t = 0; t < 16; ++t)
#pragma unroll
    for (int r = 0; r < 8; ++r)
      ldsW[(8 * h + r) * D_MODEL + t * 16 + n] = f2bf(ctx[t][r] * inv[r]);
  wave_lds_fence();

  vbf16x16 ca[8];
#pragma unroll
  for (int kc = 0; kc < 8; ++kc) ca[kc] = load_frag_a(ldsW + kc * 32, D_MODEL);

  // ---- fused output projection: logits = ctx @ Wo + bo, plus pooling dot w = logits . cv ----
  float wpart[8];
#pragma unroll
  for (int r = 0; r < 8; ++r) wpart[r] = 0.0f;

#pragma unroll 1
  for (int nt = 0; nt < 16; ++nt) {
    vbf16x16 bf[8];
#pragma unroll
    for (int kc = 0; kc < 8; ++kc)
      bf[kc] = load_frag_b(Wot + nt * 16 * D_MODEL + kc * 32, D_MODEL);
    v8f acc = {};
#pragma unroll
    for (int kc = 0; kc < 8; ++kc) acc = wmma_bf16(ca[kc], bf[kc], acc);

    const int col = nt * 16 + n;
    const float bs = bo[col];
    const float cvv = cv[col];
#pragma unroll
    for (int r = 0; r < 8; ++r) {
      float lg = acc[r] + bs;
      logits[(rowbase + q0 + 8 * h + r) * D_MODEL + col] = lg;
      wpart[r] += lg * cvv;
    }
  }
#pragma unroll
  for (int r = 0; r < 8; ++r) {
    float s = wpart[r];
    s += __shfl_xor(s, 1, 32);
    s += __shfl_xor(s, 2, 32);
    s += __shfl_xor(s, 4, 32);
    s += __shfl_xor(s, 8, 32);
    if (n == 0) wlog[bidx * SEQ + q0 + 8 * h + r] = s;
  }
}

// ---------- kernel 5: per-batch softmax over sequence dim of w ----------
__global__ void pool_softmax_kernel(const float* __restrict__ wlog, float* __restrict__ nw) {
  __shared__ float red[8];
  const int b = blockIdx.x, tid = threadIdx.x;
  const float* w = wlog + (size_t)b * SEQ;

  float mx = -3.4e38f;
  for (int i = tid; i < SEQ; i += 256) mx = fmaxf(mx, w[i]);
  for (int o = 16; o >= 1; o >>= 1) mx = fmaxf(mx, __shfl_xor(mx, o, 32));
  if ((tid & 31) == 0) red[tid >> 5] = mx;
  __syncthreads();
  float bm = red[0];
#pragma unroll
  for (int j = 1; j < 8; ++j) bm = fmaxf(bm, red[j]);
  __syncthreads();

  float s = 0.0f;
  for (int i = tid; i < SEQ; i += 256) s += __expf(w[i] - bm);
  for (int o = 16; o >= 1; o >>= 1) s += __shfl_xor(s, o, 32);
  if ((tid & 31) == 0) red[tid >> 5] = s;
  __syncthreads();
  float tot = 0.0f;
#pragma unroll
  for (int j = 0; j < 8; ++j) tot += red[j];
  const float invt = 1.0f / tot;

  for (int i = tid; i < SEQ; i += 256)
    nw[(size_t)b * SEQ + i] = __expf(w[i] - bm) * invt;
}

// ---------- kernel 6: out = logits * nw (broadcast over D) ----------
__global__ void scale_out_kernel(const float* __restrict__ logits, const float* __restrict__ nw,
                                 float* __restrict__ out, int n) {
  int i = blockIdx.x * 256 + threadIdx.x;
  if (i < n) out[i] = logits[i] * nw[i >> 8];   // i/D_MODEL
}

// ---------- host launch ----------
extern "C" void kernel_launch(void* const* d_in, const int* in_sizes, int n_in,
                              void* d_out, int out_size, void* d_ws, size_t ws_size,
                              hipStream_t stream) {
  const float* x  = (const float*)d_in[0];
  const float* Wq = (const float*)d_in[1];
  const float* bq = (const float*)d_in[2];
  const float* Wk = (const float*)d_in[3];
  const float* bk = (const float*)d_in[4];
  const float* Wv = (const float*)d_in[5];
  const float* bv = (const float*)d_in[6];
  const float* Wo = (const float*)d_in[7];
  const float* bo = (const float*)d_in[8];
  const float* cv = (const float*)d_in[9];
  float* out = (float*)d_out;

  char* ws = (char*)d_ws;
  size_t off = 0;
  auto carve = [&](size_t bytes) -> char* {
    char* p = ws + off;
    off += (bytes + 255) & ~(size_t)255;
    return p;
  };
  bf16*  xb     = (bf16*)carve((size_t)NROWS * D_MODEL * sizeof(bf16));
  bf16*  Wqt    = (bf16*)carve((size_t)D_MODEL * D_MODEL * sizeof(bf16));
  bf16*  Wkt    = (bf16*)carve((size_t)D_MODEL * D_MODEL * sizeof(bf16));
  bf16*  Wvt    = (bf16*)carve((size_t)D_MODEL * D_MODEL * sizeof(bf16));
  bf16*  Wot    = (bf16*)carve((size_t)D_MODEL * D_MODEL * sizeof(bf16));
  bf16*  Qb     = (bf16*)carve((size_t)NROWS * D_MODEL * sizeof(bf16));
  bf16*  Kb     = (bf16*)carve((size_t)NROWS * D_MODEL * sizeof(bf16));
  bf16*  VtB    = (bf16*)carve((size_t)NROWS * D_MODEL * sizeof(bf16));
  float* logits = (float*)carve((size_t)NROWS * D_MODEL * sizeof(float));
  float* wlog   = (float*)carve((size_t)BATCH * SEQ * sizeof(float));
  float* nw     = (float*)carve((size_t)BATCH * SEQ * sizeof(float));

  const int nElem = NROWS * D_MODEL;   // 8388608

  convert_x_kernel<<<(nElem + 255) / 256, 256, 0, stream>>>(x, xb, nElem);
  transpose_w_kernel<<<dim3(D_MODEL * D_MODEL / 256, 4), 256, 0, stream>>>(
      Wq, Wk, Wv, Wo, Wqt, Wkt, Wvt, Wot);
  qkv_kernel<<<dim3(NROWS / 128, 1, 3), 256, 0, stream>>>(
      xb, Wqt, Wkt, Wvt, bq, bk, bv, Qb, Kb, VtB);
  attn_kernel<<<dim3(SEQ / 128, BATCH), 256, 0, stream>>>(
      Qb, Kb, VtB, Wot, bo, cv, logits, wlog);
  pool_softmax_kernel<<<BATCH, 256, 0, stream>>>(wlog, nw);
  scale_out_kernel<<<(nElem + 255) / 256, 256, 0, stream>>>(logits, nw, out, nElem);
}
